// AIM_75144747810999
// MI455X (gfx1250) — matrix operand order repaired and checked by reading, not compiled, not run on hardware
//
#include <hip/hip_runtime.h>
#include <hip/hip_bf16.h>

#define OBS   256
#define HID   512
#define LAT   64
#define VOCAB 8192
#define NROWS 32768

typedef __attribute__((ext_vector_type(16))) _Float16 v16h;
typedef __attribute__((ext_vector_type(8)))  _Float16 v8h;
typedef __attribute__((ext_vector_type(4)))  _Float16 v4h;
typedef __attribute__((ext_vector_type(8)))  float    v8f;

static __device__ __forceinline__ v16h cat8(v8h lo, v8h hi) {
  return __builtin_shufflevector(lo, hi, 0,1,2,3,4,5,6,7,8,9,10,11,12,13,14,15);
}

static __device__ __forceinline__ v8f vzero() {
  v8f z = {0.f,0.f,0.f,0.f,0.f,0.f,0.f,0.f};
  return z;
}

// A-matrix fragment, 16x32 f16 (M x K), row-major source with stride ld (halves).
// ISA layout: lanes 0-15 = rows, kb = (lane>=16)?8:0; VGPR0-3 hold K=kb..kb+7,
// VGPR4-7 hold K=16+kb..16+kb+7.  Two contiguous 16-byte chunks per lane.
static __device__ __forceinline__ v16h ld_fragA(const _Float16* base, int ld, int lane) {
  const int r  = lane & 15;
  const int kb = (lane >> 4) << 3;
  const _Float16* p = base + r * ld + kb;
  v8h lo = *(const v8h*)(p);
  v8h hi = *(const v8h*)(p + 16);
  return cat8(lo, hi);
}

// B-matrix fragment, 32x16 f16 (K x N), sourced from a column-major (N-major)
// buffer Bt[n][k] with stride ld.  Same K order as fragA: kb = (lane>=16)?8:0;
// elements 0-7 hold K=kb..kb+7, 8-15 hold K=16+kb..16+kb+7.
static __device__ __forceinline__ v16h ld_fragB(const _Float16* base, int ld, int lane) {
  const int n  = lane & 15;
  const int ks = (lane >> 4) << 3;
  const _Float16* p = base + n * ld + ks;
  v8h lo = *(const v8h*)(p);
  v8h hi = *(const v8h*)(p + 16);
  return cat8(lo, hi);
}

static __device__ __forceinline__ v8f wmma16(v16h a, v16h b, v8f c) {
  return __builtin_amdgcn_wmma_f32_16x16x32_f16(false, a, false, b, (short)0, c, false, false);
}

static __device__ __forceinline__ float wave_sum(float v) {
  #pragma unroll
  for (int m = 1; m < 32; m <<= 1) v += __shfl_xor(v, m, 32);
  return v;
}

// ---------------------------------------------------------------------------
// Prep: weight transpose + f32->f16 convert; zero loss accumulators.
// w1t[HID][OBS], w2t[LAT][HID], embh[VOCAB][LAT], d1t[HID][LAT], d2t[OBS][HID]
// ---------------------------------------------------------------------------
__global__ __launch_bounds__(256) void k_prep(
    const float* __restrict__ enc_w1, const float* __restrict__ enc_w2,
    const float* __restrict__ emb,    const float* __restrict__ dec_w1,
    const float* __restrict__ dec_w2,
    _Float16* __restrict__ w1t, _Float16* __restrict__ w2t,
    _Float16* __restrict__ embh, _Float16* __restrict__ d1t,
    _Float16* __restrict__ d2t, float* __restrict__ accum)
{
  unsigned gid = blockIdx.x * 256u + threadIdx.x;
  if (gid < 2) accum[gid] = 0.0f;
  unsigned i = gid;
  if (i < (unsigned)(HID * OBS)) {                 // w1t
    unsigned o = i / OBS, k = i % OBS;
    w1t[i] = (_Float16)enc_w1[k * HID + o];
    return;
  }
  i -= HID * OBS;
  if (i < (unsigned)(LAT * HID)) {                 // w2t
    unsigned l = i / HID, h = i % HID;
    w2t[i] = (_Float16)enc_w2[h * LAT + l];
    return;
  }
  i -= LAT * HID;
  if (i < (unsigned)(VOCAB * LAT)) {               // embh (already N-major)
    embh[i] = (_Float16)emb[i];
    return;
  }
  i -= VOCAB * LAT;
  if (i < (unsigned)(HID * LAT)) {                 // d1t
    unsigned h = i / LAT, l = i % LAT;
    d1t[i] = (_Float16)dec_w1[l * HID + h];
    return;
  }
  i -= HID * LAT;
  if (i < (unsigned)(OBS * HID)) {                 // d2t
    unsigned o = i / HID, h = i % HID;
    d2t[i] = (_Float16)dec_w2[h * OBS + o];
    return;
  }
}

__global__ __launch_bounds__(256) void k_enorm(const float* __restrict__ emb,
                                               float* __restrict__ enorm)
{
  int r = blockIdx.x * 256 + threadIdx.x;
  if (r >= VOCAB) return;
  const float* p = emb + (size_t)r * LAT;
  float s = 0.f;
  #pragma unroll
  for (int i = 0; i < LAT; ++i) { float v = p[i]; s += v * v; }
  enorm[r] = s;
}

// ---------------------------------------------------------------------------
// Encoder: x[32,256] -> GEMM -> LayerNorm -> ReLU -> GEMM -> latent[32,64]
// Block = 32 rows (2 row-tiles), 8 wave32.  B fragments reused across both
// row-tiles to halve w1t L2 traffic.
// ---------------------------------------------------------------------------
__global__ __launch_bounds__(256) void k_encoder(
    const float* __restrict__ x, const float* __restrict__ enc_b1,
    const float* __restrict__ ln_g, const float* __restrict__ ln_b,
    const float* __restrict__ enc_b2,
    const _Float16* __restrict__ w1t, const _Float16* __restrict__ w2t,
    float* __restrict__ latf, _Float16* __restrict__ lath)
{
  __shared__ _Float16 sx[32][OBS + 8];
  __shared__ _Float16 sh[32][HID + 8];
  __shared__ float    sred[8][32][2];

  const int tid = threadIdx.x;
  const int lane = tid & 31, w = tid >> 5, hi = lane >> 4, n = lane & 15;
  const int row0 = blockIdx.x * 32;

  { // stage x tile to LDS as f16 (32 consecutive floats per thread)
    int base = tid * 32;
    int r = base >> 8, c = base & 255;
    const float* xp = x + (size_t)(row0 + r) * OBS + c;
    #pragma unroll
    for (int j = 0; j < 8; ++j) {
      float4 v = *(const float4*)(xp + j * 4);
      v4h h;
      h[0] = (_Float16)v.x; h[1] = (_Float16)v.y;
      h[2] = (_Float16)v.z; h[3] = (_Float16)v.w;
      *(v4h*)&sx[r][c + j * 4] = h;
    }
  }
  __syncthreads();

  // GEMM1: h = x @ enc_w1  (wave w owns output cols [w*64, w*64+64))
  v8f acc[2][4];
  #pragma unroll
  for (int rt = 0; rt < 2; ++rt)
    #pragma unroll
    for (int t = 0; t < 4; ++t) acc[rt][t] = vzero();
  const int colbase = w * 64;
  for (int kk = 0; kk < OBS / 32; ++kk) {
    v16h aA = ld_fragA(&sx[0][kk * 32],  OBS + 8, lane);
    v16h aB = ld_fragA(&sx[16][kk * 32], OBS + 8, lane);
    #pragma unroll
    for (int t = 0; t < 4; ++t) {
      v16h b = ld_fragB(w1t + (size_t)(colbase + t * 16) * OBS + kk * 32, OBS, lane);
      acc[0][t] = wmma16(aA, b, acc[0][t]);
      acc[1][t] = wmma16(aB, b, acc[1][t]);
    }
  }

  // bias + LayerNorm stats
  float hv[2][4][8];
  #pragma unroll
  for (int t = 0; t < 4; ++t) {
    float b1 = enc_b1[colbase + t * 16 + n];
    #pragma unroll
    for (int rt = 0; rt < 2; ++rt)
      #pragma unroll
      for (int v = 0; v < 8; ++v) hv[rt][t][v] = acc[rt][t][v] + b1;
  }
  #pragma unroll
  for (int rt = 0; rt < 2; ++rt) {
    #pragma unroll
    for (int v = 0; v < 8; ++v) {
      float s = hv[rt][0][v] + hv[rt][1][v] + hv[rt][2][v] + hv[rt][3][v];
      float q = hv[rt][0][v]*hv[rt][0][v] + hv[rt][1][v]*hv[rt][1][v]
              + hv[rt][2][v]*hv[rt][2][v] + hv[rt][3][v]*hv[rt][3][v];
      #pragma unroll
      for (int m = 1; m < 16; m <<= 1) { s += __shfl_xor(s, m, 32); q += __shfl_xor(q, m, 32); }
      if (n == 0) {
        sred[w][rt * 16 + v + 8 * hi][0] = s;
        sred[w][rt * 16 + v + 8 * hi][1] = q;
      }
    }
  }
  __syncthreads();

  // normalize + ReLU -> f16 LDS
  #pragma unroll
  for (int rt = 0; rt < 2; ++rt) {
    #pragma unroll
    for (int v = 0; v < 8; ++v) {
      int row = rt * 16 + v + 8 * hi;
      float S = 0.f, Q = 0.f;
      #pragma unroll
      for (int ww = 0; ww < 8; ++ww) { S += sred[ww][row][0]; Q += sred[ww][row][1]; }
      float mu = S * (1.0f / HID);
      float var = Q * (1.0f / HID) - mu * mu;
      float rinv = rsqrtf(var + 1e-5f);
      #pragma unroll
      for (int t = 0; t < 4; ++t) {
        int col = colbase + t * 16 + n;
        float hn = (hv[rt][t][v] - mu) * rinv * ln_g[col] + ln_b[col];
        sh[row][col] = (_Float16)fmaxf(hn, 0.0f);
      }
    }
  }
  __syncthreads();

  // GEMM2: latent = h @ enc_w2 + b2 (8 tiles: wave w -> row-tile w>>2, col-tile w&3)
  {
    const int rt = w >> 2, ct = w & 3;
    v8f a2 = vzero();
    for (int kk = 0; kk < HID / 32; ++kk) {
      v16h a = ld_fragA(&sh[rt * 16][kk * 32], HID + 8, lane);
      v16h b = ld_fragB(w2t + (size_t)(ct * 16) * HID + kk * 32, HID, lane);
      a2 = wmma16(a, b, a2);
    }
    int col = ct * 16 + n;
    float b2 = enc_b2[col];
    #pragma unroll
    for (int v = 0; v < 8; ++v) {
      int row = rt * 16 + v + 8 * hi;
      float val = a2[v] + b2;
      latf[(size_t)(row0 + row) * LAT + col] = val;
      lath[(size_t)(row0 + row) * LAT + col] = (_Float16)val;
    }
  }
}

// ---------------------------------------------------------------------------
// Codebook argmin: d2 = ||e||^2 - 2 <latent, e>   (||latent||^2 constant/row)
// Block = 64 rows (4 row-tiles in registers); each B fragment feeds 4 WMMAs.
// ---------------------------------------------------------------------------
__global__ __launch_bounds__(256) void k_codebook(
    const _Float16* __restrict__ lath, const _Float16* __restrict__ embh,
    const float* __restrict__ enorm, int* __restrict__ idx)
{
  __shared__ _Float16 slat[64][LAT];
  __shared__ float sbd[8][64];
  __shared__ int   sbj[8][64];

  const int tid = threadIdx.x;
  const int lane = tid & 31, w = tid >> 5, hi = lane >> 4, n = lane & 15;
  const int row0 = blockIdx.x * 64;

  { // 64x64 latent tile -> LDS (16 halves per thread)
    int base = tid * 16, r = base >> 6, c = base & 63;
    *(v8h*)&slat[r][c]     = *(const v8h*)(lath + (size_t)(row0 + r) * LAT + c);
    *(v8h*)&slat[r][c + 8] = *(const v8h*)(lath + (size_t)(row0 + r) * LAT + c + 8);
  }
  __syncthreads();

  v16h a0[4], a1[4];
  #pragma unroll
  for (int rt = 0; rt < 4; ++rt) {
    a0[rt] = ld_fragA(&slat[rt * 16][0],  LAT, lane);
    a1[rt] = ld_fragA(&slat[rt * 16][32], LAT, lane);
  }

  float bd[4][8]; int bj[4][8];
  #pragma unroll
  for (int rt = 0; rt < 4; ++rt)
    #pragma unroll
    for (int v = 0; v < 8; ++v) { bd[rt][v] = 3.4e38f; bj[rt][v] = 0; }

  for (int jc = 0; jc < VOCAB; jc += 128) {
    int jb = jc + w * 16;
    v16h b0 = ld_fragB(embh + (size_t)jb * LAT,      LAT, lane);
    v16h b1 = ld_fragB(embh + (size_t)jb * LAT + 32, LAT, lane);
    if (jc + 128 < VOCAB)   // prefetch next chunk of the codebook stream
      __builtin_prefetch(embh + (size_t)(jb + 128 + n) * LAT, 0, 3);
    float nrm = enorm[jb + n];
    #pragma unroll
    for (int rt = 0; rt < 4; ++rt) {
      v8f c = vzero();
      c = wmma16(a0[rt], b0, c);
      c = wmma16(a1[rt], b1, c);
      #pragma unroll
      for (int v = 0; v < 8; ++v) {
        float d = nrm - 2.0f * c[v];
        if (d < bd[rt][v]) { bd[rt][v] = d; bj[rt][v] = jb + n; }
      }
    }
  }

  #pragma unroll
  for (int rt = 0; rt < 4; ++rt) {
    #pragma unroll
    for (int v = 0; v < 8; ++v) {
      float d = bd[rt][v]; int j = bj[rt][v];
      #pragma unroll
      for (int m = 1; m < 16; m <<= 1) {
        float od = __shfl_xor(d, m, 32);
        int   oj = __shfl_xor(j, m, 32);
        if (od < d || (od == d && oj < j)) { d = od; j = oj; }
      }
      if (n == 0) {
        sbd[w][rt * 16 + v + 8 * hi] = d;
        sbj[w][rt * 16 + v + 8 * hi] = j;
      }
    }
  }
  __syncthreads();

  if (tid < 64) {
    float d = sbd[0][tid]; int j = sbj[0][tid];
    #pragma unroll
    for (int ww = 1; ww < 8; ++ww) {
      float od = sbd[ww][tid]; int oj = sbj[ww][tid];
      if (od < d || (od == d && oj < j)) { d = od; j = oj; }
    }
    idx[row0 + tid] = j;
  }
}

// ---------------------------------------------------------------------------
// Decoder + losses: q=emb[idx]; rec = relu(q@dec_w1+b1)@dec_w2+b2
// Block = 32 rows; B fragments reused across both row-tiles.
// accum[0] += sum (rec-x)^2 ; accum[1] += sum (q - latent)^2  (fp32)
// ---------------------------------------------------------------------------
__global__ __launch_bounds__(256) void k_decoder(
    const float* __restrict__ x, const float* __restrict__ latf,
    const float* __restrict__ emb, const _Float16* __restrict__ embh,
    const float* __restrict__ dec_b1, const float* __restrict__ dec_b2,
    const _Float16* __restrict__ d1t, const _Float16* __restrict__ d2t,
    const int* __restrict__ idx, float* __restrict__ accum)
{
  __shared__ _Float16 sq[32][LAT];
  __shared__ _Float16 shq[32][HID + 8];
  __shared__ int sidx[32];

  const int tid = threadIdx.x;
  const int lane = tid & 31, w = tid >> 5, hi = lane >> 4, n = lane & 15;
  const int row0 = blockIdx.x * 32;

  if (tid < 32) sidx[tid] = idx[row0 + tid];
  __syncthreads();

  { // gather quantised rows (f16) into LDS (8 halves per thread)
    int base = tid * 8, r = base >> 6, c = base & 63;
    *(v8h*)&sq[r][c] = *(const v8h*)(embh + (size_t)sidx[r] * LAT + c);
  }

  // codebook/commitment loss partial sums (fp32 inputs)
  float cb = 0.f;
  #pragma unroll
  for (int jj = 0; jj < 8; ++jj) {
    int i = tid + jj * 256;
    int r = i >> 6, c = i & 63;
    float q = emb[(size_t)sidx[r] * LAT + c];
    float l = latf[(size_t)(row0 + r) * LAT + c];
    float dif = q - l;
    cb += dif * dif;
  }
  cb = wave_sum(cb);
  if (lane == 0) atomicAdd(accum + 1, cb);
  __syncthreads();

  // GEMM d1: hq = relu(q @ dec_w1 + b1), 32x512
  v8f acc[2][4];
  #pragma unroll
  for (int rt = 0; rt < 2; ++rt)
    #pragma unroll
    for (int t = 0; t < 4; ++t) acc[rt][t] = vzero();
  #pragma unroll
  for (int kk = 0; kk < 2; ++kk) {
    v16h aA = ld_fragA(&sq[0][kk * 32],  LAT, lane);
    v16h aB = ld_fragA(&sq[16][kk * 32], LAT, lane);
    #pragma unroll
    for (int t = 0; t < 4; ++t) {
      int col0 = w * 64 + t * 16;
      v16h b = ld_fragB(d1t + (size_t)col0 * LAT + kk * 32, LAT, lane);
      acc[0][t] = wmma16(aA, b, acc[0][t]);
      acc[1][t] = wmma16(aB, b, acc[1][t]);
    }
  }
  #pragma unroll
  for (int t = 0; t < 4; ++t) {
    int col = w * 64 + t * 16 + n;
    float b1 = dec_b1[col];
    #pragma unroll
    for (int rt = 0; rt < 2; ++rt)
      #pragma unroll
      for (int v = 0; v < 8; ++v) {
        int row = rt * 16 + v + 8 * hi;
        shq[row][col] = (_Float16)fmaxf(acc[rt][t][v] + b1, 0.0f);
      }
  }
  __syncthreads();

  // GEMM d2: rec = hq @ dec_w2 + b2, 32x256 ; fused recon loss
  float rl = 0.0f;
  v8f a2[2][2];
  #pragma unroll
  for (int rt = 0; rt < 2; ++rt)
    #pragma unroll
    for (int t = 0; t < 2; ++t) a2[rt][t] = vzero();
  for (int kk = 0; kk < HID / 32; ++kk) {
    v16h aA = ld_fragA(&shq[0][kk * 32],  HID + 8, lane);
    v16h aB = ld_fragA(&shq[16][kk * 32], HID + 8, lane);
    #pragma unroll
    for (int t = 0; t < 2; ++t) {
      int col0 = w * 32 + t * 16;
      v16h b = ld_fragB(d2t + (size_t)col0 * HID + kk * 32, HID, lane);
      a2[0][t] = wmma16(aA, b, a2[0][t]);
      a2[1][t] = wmma16(aB, b, a2[1][t]);
    }
  }
  #pragma unroll
  for (int rt = 0; rt < 2; ++rt)
    #pragma unroll
    for (int t = 0; t < 2; ++t) {
      int col = w * 32 + t * 16 + n;
      float b2 = dec_b2[col];
      #pragma unroll
      for (int v = 0; v < 8; ++v) {
        int row = rt * 16 + v + 8 * hi;
        float rec = a2[rt][t][v] + b2;
        float d = rec - x[(size_t)(row0 + row) * OBS + col];
        rl += d * d;
      }
    }
  rl = wave_sum(rl);
  if (lane == 0) atomicAdd(accum + 0, rl);
}

__global__ void k_final(const float* __restrict__ accum, float* __restrict__ out) {
  if (threadIdx.x == 0 && blockIdx.x == 0) {
    float recon = accum[0] * (1.0f / ((float)NROWS * (float)OBS));
    float cb    = accum[1] * (1.0f / ((float)NROWS * (float)LAT));
    out[0] = 0.5f * recon + 1.5f * cb;   // codebook + 0.5*commitment (equal values)
  }
}

extern "C" void kernel_launch(void* const* d_in, const int* in_sizes, int n_in,
                              void* d_out, int out_size, void* d_ws, size_t ws_size,
                              hipStream_t stream) {
  (void)in_sizes; (void)n_in; (void)out_size; (void)ws_size;
  const float* x      = (const float*)d_in[0];
  const float* enc_w1 = (const float*)d_in[1];
  const float* enc_b1 = (const float*)d_in[2];
  const float* ln_g   = (const float*)d_in[3];
  const float* ln_b   = (const float*)d_in[4];
  const float* enc_w2 = (const float*)d_in[5];
  const float* enc_b2 = (const float*)d_in[6];
  const float* emb    = (const float*)d_in[7];
  const float* dec_w1 = (const float*)d_in[8];
  const float* dec_b1 = (const float*)d_in[9];
  const float* dec_w2 = (const float*)d_in[10];
  const float* dec_b2 = (const float*)d_in[11];

  char* ws = (char*)d_ws;
  _Float16* w1t  = (_Float16*)(ws + 0);          //  512*256 f16
  _Float16* w2t  = (_Float16*)(ws + 262144);     //   64*512 f16
  _Float16* embh = (_Float16*)(ws + 327680);     // 8192*64  f16
  _Float16* d1t  = (_Float16*)(ws + 1376256);    //  512*64  f16
  _Float16* d2t  = (_Float16*)(ws + 1441792);    //  256*512 f16
  float*    enorm= (float*)   (ws + 1703936);    // 8192 f32
  float*    latf = (float*)   (ws + 1736704);    // 32768*64 f32
  _Float16* lath = (_Float16*)(ws + 10125312);   // 32768*64 f16
  int*      idxb = (int*)     (ws + 14319616);   // 32768 i32
  float*    accum= (float*)   (ws + 14450688);   // 2 f32

  k_prep<<<3328, 256, 0, stream>>>(enc_w1, enc_w2, emb, dec_w1, dec_w2,
                                   w1t, w2t, embh, d1t, d2t, accum);
  k_enorm<<<VOCAB / 256, 256, 0, stream>>>(emb, enorm);
  k_encoder<<<NROWS / 32, 256, 0, stream>>>(x, enc_b1, ln_g, ln_b, enc_b2,
                                            w1t, w2t, latf, lath);
  k_codebook<<<NROWS / 64, 256, 0, stream>>>(lath, embh, enorm, idxb);
  k_decoder<<<NROWS / 32, 256, 0, stream>>>(x, latf, emb, embh, dec_b1, dec_b2,
                                            d1t, d2t, idxb, accum);
  k_final<<<1, 1, 0, stream>>>(accum, (float*)d_out);
}
